// MambaLayer_42125039239784
// MI455X (gfx1250) — matrix operand order repaired and checked by reading, not compiled, not run on hardware
//
#include <hip/hip_runtime.h>
#include <hip/hip_bf16.h>

typedef __bf16 v16bf __attribute__((ext_vector_type(16)));
typedef __bf16 v8bf  __attribute__((ext_vector_type(8)));
typedef float  v8f   __attribute__((ext_vector_type(8)));

#define D_STATE 16
#define D_CONV  4
#define D_INNER 1024
#define B_SZ    4
#define SEQ     2048
#define NROWS   (B_SZ * SEQ)   // 8192

static __device__ __forceinline__ __bf16 f2bf(float f) {
    // round-to-nearest-even fp32 -> bf16 (used in staging passes and kernel epilogues)
    unsigned u = __float_as_uint(f);
    unsigned r = (u + 0x7FFFu + ((u >> 16) & 1u)) >> 16;
    unsigned short s = (unsigned short)r;
    __bf16 o;
    __builtin_memcpy(&o, &s, sizeof(o));
    return o;
}

static __device__ __forceinline__ float sigmoidf_(float x) {
    return 1.0f / (1.0f + __expf(-x));
}
static __device__ __forceinline__ float softplusf_(float x) {
    return fmaxf(x, 0.0f) + log1pf(__expf(-fabsf(x)));
}

static __device__ __forceinline__ v16bf cat16(v8bf lo, v8bf hi) {
    return __builtin_shufflevector(lo, hi, 0, 1, 2, 3, 4, 5, 6, 7,
                                           8, 9, 10, 11, 12, 13, 14, 15);
}

// ---------------------------------------------------------------------------
// fp32 -> bf16 staging pass (4 elements / thread)
// ---------------------------------------------------------------------------
__global__ __launch_bounds__(256) void cvt_bf16(
    const float* __restrict__ src, __bf16* __restrict__ dst, int n4)
{
    int i = blockIdx.x * blockDim.x + threadIdx.x;
    if (i >= n4) return;
    float4 v = ((const float4*)src)[i];
    __bf16 o[4] = { f2bf(v.x), f2bf(v.y), f2bf(v.z), f2bf(v.w) };
    ((uint2*)dst)[i] = *(const uint2*)o;
}

// ---------------------------------------------------------------------------
// C[M,N] = A[M,K] @ W[N,K]^T, A/W bf16, C fp32 (+ optional bf16 copy Cbf).
// mode 0: plain   mode 1: softplus(acc + bias[n])
// One 32(M) x 64(N) macro-tile per wave: 2 A-frags x 4 B-frags -> 8 WMMA / K-step.
// ---------------------------------------------------------------------------
__global__ __launch_bounds__(256) void wmma_gemm_bf16(
    const __bf16* __restrict__ A, int lda,
    const __bf16* __restrict__ W,
    float* __restrict__ C, int ldc,
    int M, int N, int K,
    const float* __restrict__ bias, int mode,
    __bf16* __restrict__ Cbf)
{
    int wave = (int)((blockIdx.x * blockDim.x + threadIdx.x) >> 5);
    int lane = threadIdx.x & 31;
    int tiles_n = N >> 6;   // 64-wide
    int tiles_m = M >> 5;   // 32-tall
    if (wave >= tiles_m * tiles_n) return;   // wave-uniform: EXEC all-ones at WMMA
    int tm = wave / tiles_n;
    int tn = wave - tm * tiles_n;

    int mn   = lane & 15;   // A row within 16-tile / W row (output col) within 16-tile
    int half = lane >> 4;

    const __bf16* a0p = A + (size_t)(tm * 32 + mn) * lda;
    const __bf16* a1p = A + (size_t)(tm * 32 + 16 + mn) * lda;
    const __bf16* w0p = W + (size_t)(tn * 64 + mn) * K;
    const __bf16* w1p = w0p + (size_t)16 * K;
    const __bf16* w2p = w0p + (size_t)32 * K;
    const __bf16* w3p = w0p + (size_t)48 * K;

    v8f acc[2][4] = {};

    for (int k0 = 0; k0 < K; k0 += 32) {
        int ka = k0 + 16 * half;   // A: two adjacent 8-elem 16B chunks = 16 K-contiguous
        int kb = k0 + 16 * half;   // B: one 16-elem K-contiguous 32B chunk

        // prefetch streaming A two K-steps ahead (global_prefetch_b8)
        __builtin_prefetch(a0p + ka + 64, 0, 1);
        __builtin_prefetch(a1p + ka + 64, 0, 1);

        v16bf a0 = cat16(*(const v8bf*)(a0p + ka), *(const v8bf*)(a0p + ka + 8));
        v16bf a1 = cat16(*(const v8bf*)(a1p + ka), *(const v8bf*)(a1p + ka + 8));
        v16bf b0 = *(const v16bf*)(w0p + kb);
        v16bf b1 = *(const v16bf*)(w1p + kb);
        v16bf b2 = *(const v16bf*)(w2p + kb);
        v16bf b3 = *(const v16bf*)(w3p + kb);

        acc[0][0] = __builtin_amdgcn_wmma_f32_16x16x32_bf16(false, a0, false, b0, (short)0, acc[0][0], false, false);
        acc[0][1] = __builtin_amdgcn_wmma_f32_16x16x32_bf16(false, a0, false, b1, (short)0, acc[0][1], false, false);
        acc[0][2] = __builtin_amdgcn_wmma_f32_16x16x32_bf16(false, a0, false, b2, (short)0, acc[0][2], false, false);
        acc[0][3] = __builtin_amdgcn_wmma_f32_16x16x32_bf16(false, a0, false, b3, (short)0, acc[0][3], false, false);
        acc[1][0] = __builtin_amdgcn_wmma_f32_16x16x32_bf16(false, a1, false, b0, (short)0, acc[1][0], false, false);
        acc[1][1] = __builtin_amdgcn_wmma_f32_16x16x32_bf16(false, a1, false, b1, (short)0, acc[1][1], false, false);
        acc[1][2] = __builtin_amdgcn_wmma_f32_16x16x32_bf16(false, a1, false, b2, (short)0, acc[1][2], false, false);
        acc[1][3] = __builtin_amdgcn_wmma_f32_16x16x32_bf16(false, a1, false, b3, (short)0, acc[1][3], false, false);
    }

    #pragma unroll
    for (int i = 0; i < 2; ++i) {
        #pragma unroll
        for (int j = 0; j < 4; ++j) {
            int nn = tn * 64 + j * 16 + mn;
            #pragma unroll
            for (int r = 0; r < 8; ++r) {
                int mm = tm * 32 + i * 16 + r + 8 * half;
                float v = acc[i][j][r];
                if (mode == 1) v = softplusf_(v + bias[nn]);
                C[(size_t)mm * ldc + nn] = v;
                if (Cbf) Cbf[(size_t)mm * ldc + nn] = f2bf(v);
            }
        }
    }
}

// ---------------------------------------------------------------------------
// depthwise causal conv (k=4) + SiLU; reads u-half of xz (row stride 2048),
// writes uc fp32 (for the scan) and uc bf16 (for the x_dbl GEMM).
// ---------------------------------------------------------------------------
__global__ __launch_bounds__(256) void conv_silu(
    const float* __restrict__ xz, const float* __restrict__ Wc,
    const float* __restrict__ bc, float* __restrict__ uc,
    __bf16* __restrict__ ucbf)
{
    int idx = blockIdx.x * blockDim.x + threadIdx.x;     // row*1024 + d
    if (idx >= NROWS * D_INNER) return;
    int d    = idx & (D_INNER - 1);
    int row  = idx >> 10;
    int t    = row & (SEQ - 1);
    int base = row - t;                                   // b*SEQ
    float s = bc[d];
    #pragma unroll
    for (int k = 0; k < D_CONV; ++k) {
        int tt = t + k - (D_CONV - 1);
        if (tt >= 0)
            s += xz[(size_t)(base + tt) * 2048 + d] * Wc[d * D_CONV + k];
    }
    float r = s * sigmoidf_(s);
    uc[idx]   = r;
    ucbf[idx] = f2bf(r);
}

// ---------------------------------------------------------------------------
// Sequential selective scan, one lane per (b,d); 16 states in VGPRs.
// B_t/C_t (32 floats/step, shared across all d of a batch) double-buffered in LDS.
// Fused epilogue: y = (y_scan + u*Dp) * silu(z), written as bf16 for out-GEMM.
// ---------------------------------------------------------------------------
__global__ __launch_bounds__(128) void scan_fused(
    const float* __restrict__ delta, const float* __restrict__ uc,
    const float* __restrict__ xdbl,  const float* __restrict__ A_log,
    const float* __restrict__ Dp,    const float* __restrict__ xz,
    __bf16* __restrict__ ybf)
{
    __shared__ float bcbuf[2][32];
    int b = blockIdx.x >> 3;
    int d = ((blockIdx.x & 7) << 7) + threadIdx.x;

    float a[D_STATE], h[D_STATE];
    #pragma unroll
    for (int n = 0; n < D_STATE; ++n) {
        a[n] = -__expf(A_log[d * D_STATE + n]);
        h[n] = 0.0f;
    }
    float dp = Dp[d];

    if (threadIdx.x < 32)
        bcbuf[0][threadIdx.x] = xdbl[(size_t)(b * SEQ) * 64 + 32 + threadIdx.x];
    __syncthreads();

    for (int t = 0; t < SEQ; ++t) {
        int row = b * SEQ + t;
        int cur = t & 1;
        if (t + 1 < SEQ && threadIdx.x < 32)
            bcbuf[cur ^ 1][threadIdx.x] = xdbl[(size_t)(row + 1) * 64 + 32 + threadIdx.x];

        float dlt = delta[(size_t)row * D_INNER + d];
        float u   = uc[(size_t)row * D_INNER + d];
        float du  = dlt * u;
        float y   = 0.0f;
        #pragma unroll
        for (int n = 0; n < D_STATE; ++n) {
            h[n] = __expf(dlt * a[n]) * h[n] + du * bcbuf[cur][n];
            y += h[n] * bcbuf[cur][16 + n];
        }
        float z = xz[(size_t)row * 2048 + 1024 + d];
        ybf[(size_t)row * D_INNER + d] = f2bf((y + u * dp) * (z * sigmoidf_(z)));
        __syncthreads();
    }
}

extern "C" void kernel_launch(void* const* d_in, const int* in_sizes, int n_in,
                              void* d_out, int out_size, void* d_ws, size_t ws_size,
                              hipStream_t stream) {
    const float* x      = (const float*)d_in[0];
    const float* W_in   = (const float*)d_in[1];
    const float* W_conv = (const float*)d_in[2];
    const float* b_conv = (const float*)d_in[3];
    const float* W_x    = (const float*)d_in[4];
    const float* W_dt   = (const float*)d_in[5];
    const float* b_dt   = (const float*)d_in[6];
    const float* A_log  = (const float*)d_in[7];
    const float* Dp     = (const float*)d_in[8];
    const float* W_out  = (const float*)d_in[9];
    float* out = (float*)d_out;

    // fp32 workspace
    float* xz   = (float*)d_ws;                       // 8192 x 2048
    float* uc   = xz   + (size_t)NROWS * 2048;        // 8192 x 1024
    float* xdbl = uc   + (size_t)NROWS * 1024;        // 8192 x 64
    float* dlt  = xdbl + (size_t)NROWS * 64;          // 8192 x 1024
    // bf16 workspace
    __bf16* x_bf    = (__bf16*)(dlt + (size_t)NROWS * 1024);
    __bf16* Win_bf  = x_bf    + (size_t)NROWS * 512;      // 2048 x 512
    __bf16* Wx_bf   = Win_bf  + (size_t)2048 * 512;       // 64 x 1024
    __bf16* Wdt_bf  = Wx_bf   + (size_t)64 * 1024;        // 1024 x 32
    __bf16* Wout_bf = Wdt_bf  + (size_t)1024 * 32;        // 512 x 1024
    __bf16* uc_bf   = Wout_bf + (size_t)512 * 1024;       // 8192 x 1024
    __bf16* xdbl_bf = uc_bf   + (size_t)NROWS * 1024;     // 8192 x 64
    __bf16* y_bf    = xdbl_bf + (size_t)NROWS * 64;       // 8192 x 1024

    auto cvtBlocks = [](size_t n) { return (int)((n / 4 + 255) / 256); };
    auto gemmBlocks = [](int tiles) { return (tiles + 7) / 8; }; // 8 waves / block

    // Stage bf16 copies of x and weights
    cvt_bf16<<<cvtBlocks((size_t)NROWS * 512), 256, 0, stream>>>(x,     x_bf,    NROWS * 512 / 4);
    cvt_bf16<<<cvtBlocks((size_t)2048 * 512),  256, 0, stream>>>(W_in,  Win_bf,  2048 * 512 / 4);
    cvt_bf16<<<cvtBlocks((size_t)64 * 1024),   256, 0, stream>>>(W_x,   Wx_bf,   64 * 1024 / 4);
    cvt_bf16<<<cvtBlocks((size_t)1024 * 32),   256, 0, stream>>>(W_dt,  Wdt_bf,  1024 * 32 / 4);
    cvt_bf16<<<cvtBlocks((size_t)512 * 1024),  256, 0, stream>>>(W_out, Wout_bf, 512 * 1024 / 4);

    // 1) xz = x @ W_in^T      (M=8192, N=2048, K=512)
    wmma_gemm_bf16<<<gemmBlocks((NROWS / 32) * (2048 / 64)), 256, 0, stream>>>(
        x_bf, 512, Win_bf, xz, 2048, NROWS, 2048, 512, nullptr, 0, nullptr);

    // 2) uc = silu(conv(u) + b_conv)   (fp32 + bf16 outputs)
    conv_silu<<<(NROWS * D_INNER) / 256, 256, 0, stream>>>(xz, W_conv, b_conv, uc, uc_bf);

    // 3) x_dbl = uc @ W_x^T   (M=8192, N=64, K=1024); also emit bf16 copy
    wmma_gemm_bf16<<<gemmBlocks((NROWS / 32) * (64 / 64)), 256, 0, stream>>>(
        uc_bf, 1024, Wx_bf, xdbl, 64, NROWS, 64, 1024, nullptr, 0, xdbl_bf);

    // 4) delta = softplus(x_dbl[:, :32] @ W_dt^T + b_dt)   (M=8192, N=1024, K=32)
    wmma_gemm_bf16<<<gemmBlocks((NROWS / 32) * (1024 / 64)), 256, 0, stream>>>(
        xdbl_bf, 64, Wdt_bf, dlt, 1024, NROWS, 1024, 32, b_dt, 1, nullptr);

    // 5) selective scan + fused (y + u*Dp) * silu(z)  -> y_bf
    scan_fused<<<B_SZ * 8, 128, 0, stream>>>(dlt, uc, xdbl, A_log, Dp, xz, y_bf);

    // 6) out = y @ W_out^T    (M=8192, N=512, K=1024)
    wmma_gemm_bf16<<<gemmBlocks((NROWS / 32) * (512 / 64)), 256, 0, stream>>>(
        y_bf, 1024, Wout_bf, out, 512, NROWS, 512, 1024, nullptr, 0, nullptr);
}
